// QKVAttention_9792525435529
// MI455X (gfx1250) — matrix operand order repaired and checked by reading, not compiled, not run on hardware
//
#include <hip/hip_runtime.h>

// Flash-attention for qkv [32, 3*128, 2048] fp32 (channel-major), out [32,128,2048] fp32.
// Single pass, online softmax, f16 WMMA / f32 accumulation, gfx1250 wave32.
// S is computed TRANSPOSED (S^T = K Q^T) so the softmax reduction is in-lane,
// and O is accumulated transposed (O^T = V P^T) so alpha/l are per-lane scalars
// and the final stores are coalesced along t. K/V tiles are double-buffered in LDS.

typedef __attribute__((ext_vector_type(16))) _Float16 v16h;
typedef __attribute__((ext_vector_type(8)))  _Float16 v8h;
typedef __attribute__((ext_vector_type(2)))  _Float16 v2h;
typedef __attribute__((ext_vector_type(8)))  float    v8f;

#define NB 32
#define CH 128
#define TT 2048
#define BM 128          // query rows per workgroup (8 waves x 16)
#define BS 32           // key/value columns per iteration
#define NSB (TT / BS)   // 64 iterations
#define KT_STR 136      // kT row stride in halfs (128+8 pad) -> 272B rows, 16B aligned
#define VS_STR 40       // vS row stride in halfs (32+8 pad)  -> 80B rows,  16B aligned
#define PT_STR 40       // p  row stride in halfs

__global__ __launch_bounds__(256, 1)
void fattn_kernel(const float* __restrict__ qkv, float* __restrict__ out)
{
    const int tid  = threadIdx.x;
    const int lane = tid & 31;
    const int wave = tid >> 5;
    const int l16  = lane & 15;
    const int hi   = lane >> 4;                 // 0: lanes 0-15, 1: lanes 16-31

    const int n    = blockIdx.x >> 4;           // head
    const int mb   = blockIdx.x & 15;           // 128-query block
    const int t0w  = mb * BM + wave * 16;       // this wave's 16 query columns

    const float scale4 = 0.29730177875068026f;  // 128^-0.25, applied to q AND k

    const float* qp = qkv + (size_t)n * 3 * CH * TT;
    const float* kp = qp + (size_t)CH * TT;
    const float* vp = qp + (size_t)2 * CH * TT;

    // double-buffered K (transposed [s][c]) and V ([c][s]); per-wave P^T scratch
    __shared__ __align__(16) _Float16 kT[2][BS * KT_STR];
    __shared__ __align__(16) _Float16 vS[2][CH * VS_STR];
    __shared__ __align__(16) _Float16 pS[8 * 16 * PT_STR];
    _Float16* pw = &pS[wave * 16 * PT_STR];

    const int kb   = hi ? 8 : 0;                // A-layout K-base for this lane half
    const int koff = hi ? 16 : 0;               // unused; B fragments use kb, same k order as A

    // ---- Q: load once as B-fragments (32c x 16t), 4 k-steps over C=128 ----
    // B layout: lanes 0-15 N=t=l16, K=0..7,16..23; lanes 16-31 K=8..15,24..31 (same as A).
    v16h qB[4];
    #pragma unroll
    for (int kk = 0; kk < 4; ++kk) {
        #pragma unroll
        for (int e = 0; e < 16; ++e) {
            const int c = kk * 32 + kb + e + (e >= 8 ? 8 : 0);
            qB[kk][e] = (_Float16)(qp[(size_t)c * TT + t0w + l16] * scale4);
        }
    }

    v8f oacc[8];
    #pragma unroll
    for (int i = 0; i < 8; ++i) oacc[i] = {};
    float m_i = -1e30f, l_i = 0.0f;

    // ---- staging helpers: each thread owns 8 (c, s-pair) elements of the 128x32 tile ----
    float2 grk[8], grv[8];
    #pragma unroll
    for (int i = 0; i < 8; ++i) {                       // prologue: load tile 0
        const int f = tid + i * 256;                    // 0..2047
        const int c = f >> 4, sp = (f & 15) << 1;       // consecutive lanes -> consecutive s
        grk[i] = *(const float2*)&kp[(size_t)c * TT + sp];
        grv[i] = *(const float2*)&vp[(size_t)c * TT + sp];
    }
    #pragma unroll
    for (int i = 0; i < 8; ++i) {                       // convert + store tile 0 into buf 0
        const int f = tid + i * 256;
        const int c = f >> 4, sp = (f & 15) << 1;
        kT[0][sp * KT_STR + c]       = (_Float16)(grk[i].x * scale4);
        kT[0][(sp + 1) * KT_STR + c] = (_Float16)(grk[i].y * scale4);
        v2h hv; hv[0] = (_Float16)grv[i].x; hv[1] = (_Float16)grv[i].y;
        *(v2h*)&vS[0][c * VS_STR + sp] = hv;
    }
    __syncthreads();

    for (int sb = 0; sb < NSB; ++sb) {
        const int cur = sb & 1;

        // 1. issue next tile's global loads (latency hidden behind WMMAs below)
        if (sb + 1 < NSB) {
            const int s0n = (sb + 1) * BS;
            #pragma unroll
            for (int i = 0; i < 8; ++i) {
                const int f = tid + i * 256;
                const int c = f >> 4, sp = (f & 15) << 1;
                grk[i] = *(const float2*)&kp[(size_t)c * TT + s0n + sp];
                grv[i] = *(const float2*)&vp[(size_t)c * TT + s0n + sp];
            }
        }
        if (sb + 2 < NSB) {                             // warm L2 two tiles ahead
            const int pc = tid >> 1;
            __builtin_prefetch(kp + (size_t)pc * TT + (sb + 2) * BS, 0, 0);
            __builtin_prefetch(vp + (size_t)pc * TT + (sb + 2) * BS, 0, 0);
        }

        // 2a. S^T = (K s-rows) x (Q^T): two 16-row s tiles, K-dim 128 via 4 steps
        v8f S0 = {}, S1 = {};
        #pragma unroll
        for (int kk = 0; kk < 4; ++kk) {
            const _Float16* a0 = &kT[cur][(size_t)l16 * KT_STR + kk * 32 + kb];
            const _Float16* a1 = &kT[cur][(size_t)(16 + l16) * KT_STR + kk * 32 + kb];
            v8h x0 = *(const v8h*)a0, x1 = *(const v8h*)(a0 + 16);
            v8h y0 = *(const v8h*)a1, y1 = *(const v8h*)(a1 + 16);
            v16h A0 = __builtin_shufflevector(x0, x1, 0,1,2,3,4,5,6,7,8,9,10,11,12,13,14,15);
            v16h A1 = __builtin_shufflevector(y0, y1, 0,1,2,3,4,5,6,7,8,9,10,11,12,13,14,15);
            S0 = __builtin_amdgcn_wmma_f32_16x16x32_f16(false, A0, false, qB[kk], (short)0, S0, false, false);
            S1 = __builtin_amdgcn_wmma_f32_16x16x32_f16(false, A1, false, qB[kk], (short)0, S1, false, false);
        }

        // 2b. online softmax along s: 16 values in-lane + one cross-half exchange
        float tm[8];
        #pragma unroll
        for (int v = 0; v < 8; ++v) tm[v] = fmaxf(S0[v], S1[v]);
        #pragma unroll
        for (int v = 0; v < 4; ++v) tm[v] = fmaxf(tm[v], tm[v + 4]);
        tm[0] = fmaxf(tm[0], tm[2]); tm[1] = fmaxf(tm[1], tm[3]);
        float mx = fmaxf(tm[0], tm[1]);
        mx = fmaxf(mx, __shfl_xor(mx, 16));
        const float mnew  = fmaxf(m_i, mx);
        const float alpha = __expf(m_i - mnew);

        float pe0[8], pe1[8];
        #pragma unroll
        for (int v = 0; v < 8; ++v) {
            pe0[v] = __expf(S0[v] - mnew);
            pe1[v] = __expf(S1[v] - mnew);
        }
        // stash P^T in wave-private LDS (two b128 stores; same-wave DS is in-order)
        v8h h0, h1;
        #pragma unroll
        for (int v = 0; v < 8; ++v) { h0[v] = (_Float16)pe0[v]; h1[v] = (_Float16)pe1[v]; }
        *(v8h*)&pw[(size_t)l16 * PT_STR + 8 * hi]      = h0;   // s = 0..15 half
        *(v8h*)&pw[(size_t)l16 * PT_STR + 16 + 8 * hi] = h1;   // s = 16..31 half

        float ts[8];
        #pragma unroll
        for (int v = 0; v < 8; ++v) ts[v] = pe0[v] + pe1[v];
        #pragma unroll
        for (int v = 0; v < 4; ++v) ts[v] += ts[v + 4];
        ts[0] += ts[2]; ts[1] += ts[3];
        float ps = ts[0] + ts[1];
        ps += __shfl_xor(ps, 16);
        l_i = l_i * alpha + ps;
        m_i = mnew;

        // 2c. reload P^T as B-fragment (32s x 16t)
        const _Float16* pb = &pw[(size_t)l16 * PT_STR + kb];
        v8h p0 = *(const v8h*)pb, p1 = *(const v8h*)(pb + 16);
        v16h B_P = __builtin_shufflevector(p0, p1, 0,1,2,3,4,5,6,7,8,9,10,11,12,13,14,15);

        // 2d. O^T = alpha*O^T + (V c-rows) x P^T : 8 channel tiles, K-dim 32
        #pragma unroll
        for (int ct = 0; ct < 8; ++ct) {
            const _Float16* av = &vS[cur][(size_t)(ct * 16 + l16) * VS_STR + kb];
            v8h u0 = *(const v8h*)av, u1 = *(const v8h*)(av + 16);
            v16h A_V = __builtin_shufflevector(u0, u1, 0,1,2,3,4,5,6,7,8,9,10,11,12,13,14,15);
            v8f oc = oacc[ct];
            #pragma unroll
            for (int v = 0; v < 8; ++v) oc[v] *= alpha;
            oacc[ct] = __builtin_amdgcn_wmma_f32_16x16x32_f16(false, A_V, false, B_P, (short)0, oc, false, false);
        }

        // 3. convert + store next tile into the other buffer
        if (sb + 1 < NSB) {
            const int nxt = cur ^ 1;
            #pragma unroll
            for (int i = 0; i < 8; ++i) {
                const int f = tid + i * 256;
                const int c = f >> 4, sp = (f & 15) << 1;
                kT[nxt][sp * KT_STR + c]       = (_Float16)(grk[i].x * scale4);
                kT[nxt][(sp + 1) * KT_STR + c] = (_Float16)(grk[i].y * scale4);
                v2h hv; hv[0] = (_Float16)grv[i].x; hv[1] = (_Float16)grv[i].y;
                *(v2h*)&vS[nxt][c * VS_STR + sp] = hv;
            }
        }
        // 4. one barrier per iteration
        __syncthreads();
    }

    // ---- epilogue: per-lane normalize, directly coalesced stores (lanes sweep t) ----
    const float inv = 1.0f / l_i;
    #pragma unroll
    for (int ct = 0; ct < 8; ++ct) {
        #pragma unroll
        for (int v = 0; v < 8; ++v) {
            const int c = ct * 16 + v + 8 * hi;
            out[((size_t)n * CH + c) * TT + t0w + l16] = oacc[ct][v] * inv;
        }
    }
}

extern "C" void kernel_launch(void* const* d_in, const int* in_sizes, int n_in,
                              void* d_out, int out_size, void* d_ws, size_t ws_size,
                              hipStream_t stream)
{
    (void)in_sizes; (void)n_in; (void)out_size; (void)d_ws; (void)ws_size;
    const float* qkv = (const float*)d_in[0];
    float* out = (float*)d_out;
    dim3 grid(NB * (TT / BM));   // 512 workgroups
    dim3 block(256);             // 8 waves of 32
    fattn_kernel<<<grid, block, 0, stream>>>(qkv, out);
}
